// SelfAttention_77481210020596
// MI455X (gfx1250) — compile-verified, matrix operand order checked by reading, not run on hardware
//
#include <hip/hip_runtime.h>
#include <hip/hip_bf16.h>
#include <stdint.h>

// Problem constants (match reference)
#define B_  8
#define S_  2048
#define E_  1024
#define DK_ 64
#define H_  16

typedef __attribute__((ext_vector_type(16))) __bf16 v16bf;
typedef __attribute__((ext_vector_type(8)))  __bf16 v8bf;
typedef __attribute__((ext_vector_type(8)))  float  v8f;

// Force a fragment to be materialized/live at this point (scheduling fence via
// data dependence; expands to zero instructions).
#define TOUCH(x) asm volatile("" : "+v"(x))

// D = A(16x32 bf16) x B(32x16 bf16) + C(16x16 f32)
static __device__ __forceinline__ v8f wmma_bf16(v16bf a, v16bf b, v8f c) {
  return __builtin_amdgcn_wmma_f32_16x16x32_bf16(false, a, false, b, (short)0, c,
                                                 false, false);
}

// Assemble a 16x32 (or 32x16) bf16 fragment from two 16-byte groups.
// Fragment dword r holds K-pair (2(r%4)+8h+16(r/4), +1): regs 0..3 and 4..7 are
// each 4 consecutive dwords in memory when the K-dim is the fast axis.
static __device__ __forceinline__ v16bf frag2(const uint32_t* p0, const uint32_t* p1) {
  v8bf lo = *(const v8bf*)p0;
  v8bf hi = *(const v8bf*)p1;
  return __builtin_shufflevector(lo, hi, 0,1,2,3,4,5,6,7,8,9,10,11,12,13,14,15);
}

// Row reductions across the 16 lanes of a wave32 half (ds_swizzle xor 1/2/4/8).
static __device__ __forceinline__ float red16_max(float v) {
  int x;
  x = __builtin_amdgcn_ds_swizzle(__builtin_bit_cast(int, v), 0x041f);
  v = fmaxf(v, __builtin_bit_cast(float, x));
  x = __builtin_amdgcn_ds_swizzle(__builtin_bit_cast(int, v), 0x081f);
  v = fmaxf(v, __builtin_bit_cast(float, x));
  x = __builtin_amdgcn_ds_swizzle(__builtin_bit_cast(int, v), 0x101f);
  v = fmaxf(v, __builtin_bit_cast(float, x));
  x = __builtin_amdgcn_ds_swizzle(__builtin_bit_cast(int, v), 0x201f);
  v = fmaxf(v, __builtin_bit_cast(float, x));
  return v;
}
static __device__ __forceinline__ float red16_sum(float v) {
  int x;
  x = __builtin_amdgcn_ds_swizzle(__builtin_bit_cast(int, v), 0x041f);
  v += __builtin_bit_cast(float, x);
  x = __builtin_amdgcn_ds_swizzle(__builtin_bit_cast(int, v), 0x081f);
  v += __builtin_bit_cast(float, x);
  x = __builtin_amdgcn_ds_swizzle(__builtin_bit_cast(int, v), 0x101f);
  v += __builtin_bit_cast(float, x);
  x = __builtin_amdgcn_ds_swizzle(__builtin_bit_cast(int, v), 0x201f);
  v += __builtin_bit_cast(float, x);
  return v;
}

static __device__ __forceinline__ uint32_t pack_bf(float lo, float hi) {
  union { __bf16 h[2]; uint32_t u; } p;
  p.h[0] = (__bf16)lo; p.h[1] = (__bf16)hi;
  return p.u;
}

// Async global -> LDS copy of 16 bytes per lane (CDNA5 GLOBAL_LOAD_ASYNC_TO_LDS_B128).
// ldsoff = LDS byte offset (low 32 bits of a generic shared pointer, per ISA 10.2).
static __device__ __forceinline__ void async_copy_b128(uint32_t ldsoff, const void* gaddr) {
  asm volatile("global_load_async_to_lds_b128 %0, %1, off"
               :: "v"(ldsoff), "v"(gaddr)
               : "memory");
}

// One matrix group: 4 B fragments from LDS (8 ds_load_b128 batched, partial
// waits), then 4 WMMAs. Keeps only 32 extra VGPRs live at a time.
static __device__ __forceinline__ void gemm_group(const uint32_t* base, int lane,
                                                  v16bf a, v8f acc[4]) {
  v16bf f0 = *(const v16bf*)(base + (0 * 32 + lane) * 8);
  v16bf f1 = *(const v16bf*)(base + (1 * 32 + lane) * 8);
  v16bf f2 = *(const v16bf*)(base + (2 * 32 + lane) * 8);
  v16bf f3 = *(const v16bf*)(base + (3 * 32 + lane) * 8);
  TOUCH(f0); TOUCH(f1); TOUCH(f2); TOUCH(f3);
  acc[0] = wmma_bf16(a, f0, acc[0]);
  acc[1] = wmma_bf16(a, f1, acc[1]);
  acc[2] = wmma_bf16(a, f2, acc[2]);
  acc[3] = wmma_bf16(a, f3, acc[3]);
}

// ---------------------------------------------------------------------------
// 1) x [B,E,S] f32 -> xt [B,S,E] bf16 (transpose + downconvert)
__global__ void k_x_transpose_bf16(const float* __restrict__ x,
                                   __bf16* __restrict__ xt) {
  size_t t = (size_t)blockIdx.x * 256 + threadIdx.x;   // coalesced read (s fastest)
  int s = (int)(t % S_);
  int e = (int)((t / S_) % E_);
  int b = (int)(t / ((size_t)S_ * E_));
  xt[((size_t)(b * S_ + s)) * E_ + e] = (__bf16)x[t];
}

// ---------------------------------------------------------------------------
// 2) Swizzle a [E,DK] f32 weight (head 1 slice) into B-fragment order:
//    dword index = ((kb*4+nt)*32 + lane)*8 + r ; value = pair (W[e0][n], W[e0+1][n])
__global__ void k_w_swizzle(const float* __restrict__ w, uint32_t* __restrict__ dst) {
  int t  = blockIdx.x * 256 + threadIdx.x;             // 32768 threads
  int r  = t & 7, l = (t >> 3) & 31, nt = (t >> 8) & 3, kb = t >> 10;
  int h  = l >> 4;
  int n  = nt * 16 + (l & 15);
  int e0 = kb * 32 + 2 * (r & 3) + 8 * h + 16 * (r >> 2);
  dst[t] = pack_bf(w[e0 * DK_ + n], w[(e0 + 1) * DK_ + n]);
}

// 2b) WoSum[d,e] = sum_h Wo[h*64+d, e], stored swizzled:
//     dword index = ((kb*64+nt)*32 + lane)*8 + r ; K-dim = d (64), N = e (1024)
__global__ void k_wo_swizzle(const float* __restrict__ wo, uint32_t* __restrict__ dst) {
  int t  = blockIdx.x * 256 + threadIdx.x;             // 32768 threads
  int r  = t & 7, l = (t >> 3) & 31, nt = (t >> 8) & 63, kb = t >> 14;
  int h  = l >> 4;
  int n  = nt * 16 + (l & 15);
  int d0 = kb * 32 + 2 * (r & 3) + 8 * h + 16 * (r >> 2);
  float lo = 0.f, hi = 0.f;
#pragma unroll
  for (int hh = 0; hh < H_; ++hh) {
    lo += wo[(size_t)(hh * DK_ + d0)     * E_ + n];
    hi += wo[(size_t)(hh * DK_ + d0 + 1) * E_ + n];
  }
  dst[t] = pack_bf(lo, hi);
}

// ---------------------------------------------------------------------------
// 3) QKV projection: per wave a 16(s) x 64(dk) tile for each of Q/K/V.
//    Weight fragment blocks (3 x 4KB per k-step) are staged in LDS with
//    double-buffered async global->LDS copies shared by all 8 waves.
//    256 threads = 8 waves on 4 SIMDs -> only 2 waves/EU needed; tell the
//    compiler so it doesn't cap VGPRs (prevents accumulator spills).
__global__ void __launch_bounds__(256, 2)
k_qkv(const __bf16* __restrict__ xt,
      const uint32_t* __restrict__ wq,
      const uint32_t* __restrict__ wk,
      const uint32_t* __restrict__ wv,
      __bf16* __restrict__ Q, __bf16* __restrict__ K,
      __bf16* __restrict__ V) {
  __shared__ __align__(128) uint32_t wbuf[2][3][1024];  // 2 x 12KB weight stage
  int tid  = threadIdx.x;
  int wave = tid >> 5, lane = tid & 31;
  int b = blockIdx.x >> 4, sblk = blockIdx.x & 15;
  int s0 = sblk * 128 + wave * 16;
  int m = lane & 15, h = lane >> 4;

  const uint32_t* xrow = (const uint32_t*)xt + (size_t)(b * S_ + s0 + m) * (E_ / 2);

  uint32_t l0 = (uint32_t)(size_t)&wbuf[0][0][tid * 4];
  uint32_t l1 = (uint32_t)(size_t)&wbuf[0][1][tid * 4];
  uint32_t l2 = (uint32_t)(size_t)&wbuf[0][2][tid * 4];
  uint32_t lstep = (uint32_t)((size_t)&wbuf[1][0][0] - (size_t)&wbuf[0][0][0]);

  async_copy_b128(l0, wq + tid * 4);
  async_copy_b128(l1, wk + tid * 4);
  async_copy_b128(l2, wv + tid * 4);

  v8f aq[4] = {}, ak[4] = {}, av[4] = {};
  v16bf a_cur = frag2(xrow + 4 * h, xrow + 8 + 4 * h);
  for (int kbk = 0; kbk < 32; ++kbk) {
    int pb = kbk & 1;
    if (kbk + 1 < 32) {
      size_t gsoff = (size_t)(kbk + 1) * 1024 + tid * 4;
      uint32_t loff = (pb ? 0u : lstep);
      async_copy_b128(l0 + loff, wq + gsoff);
      async_copy_b128(l1 + loff, wk + gsoff);
      async_copy_b128(l2 + loff, wv + gsoff);
      asm volatile("s_wait_asynccnt 0x3" ::: "memory");  // current batch done
    } else {
      asm volatile("s_wait_asynccnt 0x0" ::: "memory");
    }
    __syncthreads();  // all waves' stage for kbk complete

    // fetch next iteration's A fragment early (overlaps the WMMA bursts)
    v16bf a_next = a_cur;
    if (kbk + 1 < 32)
      a_next = frag2(xrow + (kbk + 1) * 16 + 4 * h, xrow + (kbk + 1) * 16 + 8 + 4 * h);

    // three matrix groups; each batches 8 ds_loads then runs 4 WMMAs
    gemm_group(&wbuf[pb][0][0], lane, a_cur, aq);
    gemm_group(&wbuf[pb][1][0], lane, a_cur, ak);
    gemm_group(&wbuf[pb][2][0], lane, a_cur, av);

    a_cur = a_next;
    __syncthreads();  // protect buf[pb] before it is overwritten at kbk+2
  }
#pragma unroll
  for (int nt = 0; nt < 4; ++nt)
#pragma unroll
    for (int j = 0; j < 8; ++j) {
      size_t o = (size_t)(b * S_ + s0 + j + 8 * h) * DK_ + nt * 16 + m;
      Q[o] = (__bf16)aq[nt][j];
      K[o] = (__bf16)ak[nt][j];
      V[o] = (__bf16)av[nt][j];
    }
}

// ---------------------------------------------------------------------------
// 4) Repack V [B,S,64] into B-fragment order (K-dim = t): per batch,
//    dword index = ((tb*4+nt)*32 + lane)*8 + r
__global__ void k_v_swizzle(const __bf16* __restrict__ V, uint32_t* __restrict__ dst) {
  int t = blockIdx.x * 256 + threadIdx.x;              // B*65536 threads
  int b = t >> 16;
  int u = t & 65535;
  int r = u & 7, l = (u >> 3) & 31, nt = (u >> 8) & 3, tb = u >> 10;
  int h = l >> 4;
  int d  = nt * 16 + (l & 15);
  int t0 = tb * 32 + 2 * (r & 3) + 8 * h + 16 * (r >> 2);
  const __bf16* vp = V + (size_t)(b * S_ + t0) * DK_ + d;
  union { __bf16 hh[2]; uint32_t uu; } p;
  p.hh[0] = vp[0]; p.hh[1] = vp[DK_];
  dst[t] = p.uu;
}

// ---------------------------------------------------------------------------
// 5) Flash attention: per wave 16 query rows, streaming 32 keys / iteration.
//    K fragments are software-pipelined one tile ahead so their global-load
//    latency hides under the softmax VALU chain.
static __device__ __forceinline__ void load_kfrags(const uint32_t* Kdw, int t0,
                                                   int m, int h, v16bf kf[4]) {
  const uint32_t* kr0 = Kdw + (size_t)(t0 + m) * 32;
  const uint32_t* kr1 = Kdw + (size_t)(t0 + 16 + m) * 32;
  kf[0] = frag2(kr0 + 4 * h,      kr0 + 8 + 4 * h);   // d 0..31,  keys t0..t0+15
  kf[1] = frag2(kr1 + 4 * h,      kr1 + 8 + 4 * h);   // d 0..31,  keys t0+16..+31
  kf[2] = frag2(kr0 + 16 + 4 * h, kr0 + 24 + 4 * h);  // d 32..63, keys t0..t0+15
  kf[3] = frag2(kr1 + 16 + 4 * h, kr1 + 24 + 4 * h);  // d 32..63, keys t0+16..+31
}

__global__ void __launch_bounds__(256, 2)
k_attn(const __bf16* __restrict__ Q, const __bf16* __restrict__ K,
       const uint32_t* __restrict__ Vswz, __bf16* __restrict__ Hd) {
  __shared__ __align__(128) __bf16 Pb[8][512];         // per-wave 16x32 bf16 P tile
  int wave = threadIdx.x >> 5, lane = threadIdx.x & 31;
  int b = blockIdx.x >> 4, sblk = blockIdx.x & 15;
  int s0 = sblk * 128 + wave * 16;
  int m = lane & 15, h = lane >> 4;

  const uint32_t* qrow = (const uint32_t*)Q + (size_t)(b * S_ + s0 + m) * (DK_ / 2);
  v16bf qa0 = frag2(qrow + 4 * h,      qrow + 8 + 4 * h);
  v16bf qa1 = frag2(qrow + 16 + 4 * h, qrow + 24 + 4 * h);

  const uint32_t* Kdw = (const uint32_t*)K + (size_t)(b * S_) * (DK_ / 2);
  const uint32_t* Vb  = Vswz + (size_t)b * 65536;
  uint32_t* prow = (uint32_t*)&Pb[wave][0];

  v8f acc[4] = {};
  float rm[8], rs[8];
#pragma unroll
  for (int j = 0; j < 8; ++j) { rm[j] = -1e30f; rs[j] = 0.f; }

  v16bf kf[4];
  load_kfrags(Kdw, 0, m, h, kf);

  for (int tb = 0; tb < 64; ++tb) {
    // scores: two 16x16 tiles (keys t0..t0+15 and t0+16..t0+31)
    v8f c0 = {}, c1 = {};
    c0 = wmma_bf16(qa0, kf[0], c0);
    c1 = wmma_bf16(qa0, kf[1], c1);
    c0 = wmma_bf16(qa1, kf[2], c0);
    c1 = wmma_bf16(qa1, kf[3], c1);

    // pipeline: next K fragments + current V fragments overlap the softmax
    if (tb + 1 < 64) load_kfrags(Kdw, (tb + 1) * 32, m, h, kf);
    v16bf vb[4];
#pragma unroll
    for (int nt = 0; nt < 4; ++nt)
      vb[nt] = *(const v16bf*)(Vb + ((size_t)(tb * 4 + nt) * 32 + lane) * 8);
    if (tb + 2 < 64) {  // cache prefetch two tiles ahead
      __builtin_prefetch((const char*)(Kdw + (size_t)((tb + 2) * 32 + m) * 32), 0, 3);
      __builtin_prefetch((const char*)(Vb + ((size_t)((tb + 2) * 4) * 32 + lane) * 8), 0, 3);
    }

    // online softmax per row (row j+8h lives across the 16 lanes of half h)
#pragma unroll
    for (int j = 0; j < 8; ++j) {
      float a0 = c0[j] * 0.125f;   // 1/sqrt(64)
      float a1 = c1[j] * 0.125f;
      float bm = red16_max(fmaxf(a0, a1));
      float nm = fmaxf(rm[j], bm);
      float al = __expf(rm[j] - nm);
      rm[j] = nm;
      float p0 = __expf(a0 - nm);
      float p1 = __expf(a1 - nm);
      float sm = red16_sum(p0 + p1);
      rs[j] = rs[j] * al + sm;
#pragma unroll
      for (int nt = 0; nt < 4; ++nt) acc[nt][j] *= al;
      Pb[wave][(j + 8 * h) * 32 + m]      = (__bf16)p0;
      Pb[wave][(j + 8 * h) * 32 + 16 + m] = (__bf16)p1;
    }
    // P (C-layout) -> A-layout through LDS (DS ops are in-order per wave)
    v16bf pa = frag2(prow + m * 16 + 4 * h, prow + m * 16 + 8 + 4 * h);
#pragma unroll
    for (int nt = 0; nt < 4; ++nt)
      acc[nt] = wmma_bf16(pa, vb[nt], acc[nt]);
  }
  // normalize and store head (bf16, [B,S,64])
#pragma unroll
  for (int j = 0; j < 8; ++j) {
    float inv = 1.0f / rs[j];
#pragma unroll
    for (int nt = 0; nt < 4; ++nt)
      Hd[(size_t)(b * S_ + s0 + j + 8 * h) * DK_ + nt * 16 + m] =
          (__bf16)(acc[nt][j] * inv);
  }
}

// ---------------------------------------------------------------------------
// 6) out = head @ WoSum, stored transposed as [B,E,S] f32.
__global__ void __launch_bounds__(256, 2)
k_outproj(const __bf16* __restrict__ Hd,
          const uint32_t* __restrict__ woswz,
          float* __restrict__ out) {
  int wave = threadIdx.x >> 5, lane = threadIdx.x & 31;
  int gw = blockIdx.x * 8 + wave;                      // 16384 wave tiles
  int et = gw & 15;
  int st = (gw >> 4) & 127;
  int b  = gw >> 11;
  int s0 = st * 16;
  int m = lane & 15, h = lane >> 4;

  const uint32_t* hrow = (const uint32_t*)Hd + (size_t)(b * S_ + s0 + m) * (DK_ / 2);
  // preload both A fragments and all 8 Wo fragments, then a clean WMMA burst
  v16bf a0 = frag2(hrow + 4 * h,      hrow + 8 + 4 * h);
  v16bf a1 = frag2(hrow + 16 + 4 * h, hrow + 24 + 4 * h);
  v16bf bw[8];
#pragma unroll
  for (int kbk = 0; kbk < 2; ++kbk)
#pragma unroll
    for (int nt = 0; nt < 4; ++nt) {
      int col = et * 4 + nt;                           // e-tile 0..63
      bw[kbk * 4 + nt] =
          *(const v16bf*)(woswz + ((size_t)(kbk * 64 + col) * 32 + lane) * 8);
    }
#pragma unroll
  for (int i = 0; i < 8; ++i) TOUCH(bw[i]);
  v8f acc[4] = {};
#pragma unroll
  for (int nt = 0; nt < 4; ++nt) {
    acc[nt] = wmma_bf16(a0, bw[nt],     acc[nt]);
    acc[nt] = wmma_bf16(a1, bw[4 + nt], acc[nt]);
  }
#pragma unroll
  for (int nt = 0; nt < 4; ++nt)
#pragma unroll
    for (int j = 0; j < 8; ++j) {
      int e = et * 64 + nt * 16 + m;
      int s = s0 + j + 8 * h;
      out[((size_t)(b * E_ + e)) * S_ + s] = acc[nt][j];
    }
}

// ---------------------------------------------------------------------------
extern "C" void kernel_launch(void* const* d_in, const int* in_sizes, int n_in,
                              void* d_out, int out_size, void* d_ws, size_t ws_size,
                              hipStream_t stream) {
  (void)in_sizes; (void)n_in; (void)out_size; (void)ws_size;
  const float* x   = (const float*)d_in[0];
  const float* Wqs = (const float*)d_in[1];
  const float* Wks = (const float*)d_in[2];
  const float* Wvs = (const float*)d_in[3];
  const float* Wo  = (const float*)d_in[4];
  float* out = (float*)d_out;

  char* ws = (char*)d_ws;
  size_t off = 0;
  __bf16* xt = (__bf16*)(ws + off);  off += (size_t)B_ * S_ * E_ * 2;   // 32 MB
  __bf16* Q  = (__bf16*)(ws + off);  off += (size_t)B_ * S_ * DK_ * 2;  // 2 MB
  __bf16* K  = (__bf16*)(ws + off);  off += (size_t)B_ * S_ * DK_ * 2;
  __bf16* V  = (__bf16*)(ws + off);  off += (size_t)B_ * S_ * DK_ * 2;
  uint32_t* Vswz = (uint32_t*)(ws + off); off += (size_t)B_ * 65536 * 4; // 2 MB
  __bf16* Hd = (__bf16*)(ws + off);  off += (size_t)B_ * S_ * DK_ * 2;
  uint32_t* WqSwz = (uint32_t*)(ws + off); off += 32768 * 4;
  uint32_t* WkSwz = (uint32_t*)(ws + off); off += 32768 * 4;
  uint32_t* WvSwz = (uint32_t*)(ws + off); off += 32768 * 4;
  uint32_t* WoSwz = (uint32_t*)(ws + off); off += 32768 * 4;

  const size_t head1 = (size_t)1 * E_ * DK_;  // reference bug: all heads use index 1

  k_x_transpose_bf16<<<(B_ * E_ * S_) / 256, 256, 0, stream>>>(x, xt);
  k_w_swizzle<<<128, 256, 0, stream>>>(Wqs + head1, WqSwz);
  k_w_swizzle<<<128, 256, 0, stream>>>(Wks + head1, WkSwz);
  k_w_swizzle<<<128, 256, 0, stream>>>(Wvs + head1, WvSwz);
  k_wo_swizzle<<<128, 256, 0, stream>>>(Wo, WoSwz);
  k_qkv<<<B_ * (S_ / 128), 256, 0, stream>>>(xt, WqSwz, WkSwz, WvSwz, Q, K, V);
  k_v_swizzle<<<(B_ * 65536) / 256, 256, 0, stream>>>(V, Vswz);
  k_attn<<<B_ * (S_ / 128), 256, 0, stream>>>(Q, K, Vswz, Hd);
  k_outproj<<<(B_ * (S_ / 16) * (E_ / 64)) / 8, 256, 0, stream>>>(Hd, WoSwz, out);
}
